// MLPEmbedding_23785528885488
// MI455X (gfx1250) — compile-verified, matrix operand order checked by reading, not run on hardware
//
#include <hip/hip_runtime.h>

// ---------------------------------------------------------------------------
// MLP embedding update for MI455X (gfx1250, wave32, WMMA bf16 16x16x32).
//   x = memory[nodes]; h = leaky(x@W1^T + b1); y = h@W2^T + b2
//   out = memory with rows[nodes] replaced by y
// Workspace requirement: 2 * 1024*2048 bf16 = 8 MiB in d_ws (bf16 weights).
// Streaming traffic (memory copy, X gather, Y scatter) uses non-temporal
// hints so the bf16 weights stay resident in the 192MB L2.
// ---------------------------------------------------------------------------

#define FDIM   2048
#define HDIM   1024
#define NSEL   65536
#define MTILE  32
#define XPITCH (FDIM + 8)   // bf16 elems; keeps 16B alignment, kills bank conflicts
#define HPITCH (HDIM + 8)
#define YPITCH 68           // f32 elems; 272B row stride -> bank-spread reads
#define NEG_SLOPE 0.01f

typedef __bf16 bf16_t;
typedef __attribute__((ext_vector_type(4)))  __bf16 v4bf;
typedef __attribute__((ext_vector_type(8)))  __bf16 v8bf;
typedef __attribute__((ext_vector_type(16))) __bf16 v16bf;
typedef __attribute__((ext_vector_type(8)))  float  v8f;
typedef __attribute__((ext_vector_type(4)))  float  v4f;

// 16-element bf16 WMMA fragment: this lane's first 8 K-values contiguous at p,
// next 8 at p+16 elements (CDNA5 16-bit A/B layout).  Two b128 loads.
__device__ __forceinline__ v16bf frag16(const bf16_t* p) {
  v8bf lo = *(const v8bf*)(p);
  v8bf hi = *(const v8bf*)(p + 16);
  return __builtin_shufflevector(lo, hi, 0,1,2,3,4,5,6,7,8,9,10,11,12,13,14,15);
}

__device__ __forceinline__ v8f wmma_bf16(v16bf a, v16bf b, v8f c) {
  return __builtin_amdgcn_wmma_f32_16x16x32_bf16(
      false, a, false, b, (short)0, c, false, false);
}

// 16x64 output tile GEMM inner loop: 4 N-blocks, shared A fragment,
// ping-pong double-buffered K loop (unroll 2); loads for kc+2/kc+3 are issued
// half an iteration (4 WMMAs) ahead of their consuming WMMA.  Residual
// register copies co-execute on the VALU pipe while WMMA runs on the matrix
// pipe (16-bit WMMA is TRANS-class on CDNA5).
//   NK = K/32 chunks, BP = B-matrix row pitch in bf16 elements.
template <int NK, int BP>
__device__ __forceinline__ void gemm4(const bf16_t* __restrict__ aB,
                                      const bf16_t* __restrict__ bB,
                                      v8f acc[4]) {
  v16bf a0 = frag16(aB);
  v16bf a1 = frag16(aB + 32);
  v16bf b0[4], b1[4];
#pragma unroll
  for (int q = 0; q < 4; ++q) b0[q] = frag16(bB + (size_t)q * 16 * BP);
#pragma unroll
  for (int q = 0; q < 4; ++q) b1[q] = frag16(bB + (size_t)q * 16 * BP + 32);

#pragma unroll 1
  for (int kc = 0; kc < NK - 2; kc += 2) {
    // consume buffer 0, refill from kc+2
#pragma unroll
    for (int q = 0; q < 4; ++q) acc[q] = wmma_bf16(a0, b0[q], acc[q]);
    a0 = frag16(aB + (kc + 2) * 32);
#pragma unroll
    for (int q = 0; q < 4; ++q)
      b0[q] = frag16(bB + (size_t)q * 16 * BP + (kc + 2) * 32);
    // consume buffer 1, refill from kc+3
#pragma unroll
    for (int q = 0; q < 4; ++q) acc[q] = wmma_bf16(a1, b1[q], acc[q]);
    a1 = frag16(aB + (kc + 3) * 32);
#pragma unroll
    for (int q = 0; q < 4; ++q)
      b1[q] = frag16(bB + (size_t)q * 16 * BP + (kc + 3) * 32);
  }
#pragma unroll
  for (int q = 0; q < 4; ++q) acc[q] = wmma_bf16(a0, b0[q], acc[q]);
#pragma unroll
  for (int q = 0; q < 4; ++q) acc[q] = wmma_bf16(a1, b1[q], acc[q]);
}

// ---------------------------------------------------------------- weights ---
__global__ void __launch_bounds__(256)
convert_w(const v4f* __restrict__ src, bf16_t* __restrict__ dst, int n4) {
  int i = blockIdx.x * 256 + threadIdx.x;
  int stride = gridDim.x * 256;
  for (; i < n4; i += stride) {
    v4f v = __builtin_nontemporal_load(src + i);   // f32 weights read once
    v4bf o;
    o[0] = (bf16_t)v[0]; o[1] = (bf16_t)v[1]; o[2] = (bf16_t)v[2]; o[3] = (bf16_t)v[3];
    ((v4bf*)dst)[i] = o;                           // bf16 weights: keep RT (hot in L2)
  }
}

// ------------------------------------------------------------------- copy ---
__global__ void __launch_bounds__(256)
copy_mem(const v4f* __restrict__ src, v4f* __restrict__ dst, long long n4) {
  long long i = (long long)blockIdx.x * 256 + threadIdx.x;
  long long stride = (long long)gridDim.x * 256;
  for (; i < n4; i += stride) {
    v4f v = __builtin_nontemporal_load(src + i);   // pure stream: don't pollute L2
    __builtin_nontemporal_store(v, dst + i);
  }
}

// -------------------------------------------------------------- fused MLP ---
__global__ void __launch_bounds__(256)
mlp_fused(const float* __restrict__ mem, const long long* __restrict__ nodes,
          const bf16_t* __restrict__ W1b, const float* __restrict__ b1,
          const bf16_t* __restrict__ W2b, const float* __restrict__ b2,
          float* __restrict__ out) {
  extern __shared__ char smem[];
  bf16_t*    ldsX    = (bf16_t*)smem;                               // 32 x 2056 bf16
  bf16_t*    ldsH    = (bf16_t*)(smem + MTILE * XPITCH * 2);        // 32 x 1032 bf16
  long long* ldsRows = (long long*)(smem + MTILE * XPITCH * 2 + MTILE * HPITCH * 2);

  const int tid  = threadIdx.x;
  const int lane = tid & 31;
  const int w    = tid >> 5;       // wave 0..7
  const int base = blockIdx.x * MTILE;

  // ---- stage node indices ----
  if (tid < MTILE) ldsRows[tid] = nodes[base + tid];
  __syncthreads();

  // ---- gather X tile (f32 -> bf16 into LDS); read-once -> non-temporal ----
  for (int c = tid; c < MTILE * (FDIM / 4); c += 256) {
    int row  = c >> 9;             // / (FDIM/4 = 512)
    int col4 = c & 511;
    v4f v = __builtin_nontemporal_load(
        (const v4f*)(mem + (size_t)ldsRows[row] * FDIM) + col4);
    bf16_t* d = ldsX + row * XPITCH + col4 * 4;
    d[0] = (bf16_t)v[0]; d[1] = (bf16_t)v[1]; d[2] = (bf16_t)v[2]; d[3] = (bf16_t)v[3];
  }
  __syncthreads();

  const int mt   = w & 1;          // M sub-block (2 x 16 rows)
  const int nq   = w >> 1;         // N wave-group (4 waves across N)
  const int r    = lane & 15;      // row (A) / column (B,C,D) within tile
  const int half = lane >> 4;      // K-half for A/B; M-half for C/D

  // ---- phase 1: H = leaky(X @ W1^T + b1), bf16 into LDS ----
  const bf16_t* aB1 = ldsX + (mt * 16 + r) * XPITCH + half * 8;
  for (int i = 0; i < 4; ++i) {
    const int nb = i * 16 + nq * 4;                     // n-blocks nb..nb+3
    const bf16_t* bB = W1b + (size_t)(nb * 16 + r) * FDIM + half * 8;
    v8f acc[4];
#pragma unroll
    for (int q = 0; q < 4; ++q) acc[q] = (v8f){0.f,0.f,0.f,0.f,0.f,0.f,0.f,0.f};

    gemm4<FDIM / 32, FDIM>(aB1, bB, acc);

#pragma unroll
    for (int q = 0; q < 4; ++q) {
      float bias = b1[(nb + q) * 16 + r];
#pragma unroll
      for (int j = 0; j < 8; ++j) {
        float h = acc[q][j] + bias;
        h = (h >= 0.f) ? h : NEG_SLOPE * h;
        ldsH[(mt * 16 + half * 8 + j) * HPITCH + (nb + q) * 16 + r] = (bf16_t)h;
      }
    }
  }
  __syncthreads();

  // ---- phase 2: Y = H @ W2^T + b2; transpose-stage in LDS, coalesced store ----
  // ldsX region is dead now: reuse as per-wave 16x64 f32 staging tiles.
  float* ldsY = (float*)smem + (size_t)w * 16 * YPITCH;
  const int srow  = lane >> 1;                          // 0..15: output row in tile
  const int shalf = lane & 1;                           // which 32-col half
  const long long outRow = ldsRows[mt * 16 + srow];

  const bf16_t* aB2 = ldsH + (mt * 16 + r) * HPITCH + half * 8;
  for (int i = 0; i < 8; ++i) {
    const int nb = i * 16 + nq * 4;                     // n-blocks nb..nb+3
    const bf16_t* bB = W2b + (size_t)(nb * 16 + r) * HDIM + half * 8;
    v8f acc[4];
#pragma unroll
    for (int q = 0; q < 4; ++q) acc[q] = (v8f){0.f,0.f,0.f,0.f,0.f,0.f,0.f,0.f};

    gemm4<HDIM / 32, HDIM>(aB2, bB, acc);

    // stage this wave's 16x64 f32 tile (DS ops are in-order within a wave)
#pragma unroll
    for (int q = 0; q < 4; ++q) {
      float bias = b2[(nb + q) * 16 + r];
#pragma unroll
      for (int j = 0; j < 8; ++j)
        ldsY[(half * 8 + j) * YPITCH + q * 16 + r] = acc[q][j] + bias;
    }
    __builtin_amdgcn_wave_barrier();   // keep DS writes before DS reads

    // coalesced write-out: 256B contiguous per row; write-once -> non-temporal
    const long long colbase = (long long)nb * 16;
#pragma unroll
    for (int t = 0; t < 8; ++t) {
      v4f v = *(const v4f*)(ldsY + srow * YPITCH + shalf * 32 + t * 4);
      __builtin_nontemporal_store(
          v, (v4f*)(out + outRow * FDIM + colbase + shalf * 32 + t * 4));
    }
    __builtin_amdgcn_wave_barrier();   // keep reads before next iter's writes
  }
}

// ------------------------------------------------------------------ launch --
extern "C" void kernel_launch(void* const* d_in, const int* in_sizes, int n_in,
                              void* d_out, int out_size, void* d_ws, size_t ws_size,
                              hipStream_t stream) {
  const float*     mem   = (const float*)d_in[0];      // (100000, 2048) f32
  const long long* nodes = (const long long*)d_in[1];  // (65536,) int64
  const float*     W1    = (const float*)d_in[2];      // (1024, 2048) f32
  const float*     b1    = (const float*)d_in[3];      // (1024,) f32
  const float*     W2    = (const float*)d_in[4];      // (2048, 1024) f32
  const float*     b2    = (const float*)d_in[5];      // (2048,) f32
  float*           out   = (float*)d_out;              // (100000, 2048) f32

  bf16_t* W1b = (bf16_t*)d_ws;                         // 1024*2048 bf16
  bf16_t* W2b = W1b + (size_t)HDIM * FDIM;             // 2048*1024 bf16 (8 MiB total)

  const int wn4 = HDIM * FDIM / 4;                     // 524288 float4 per matrix
  convert_w<<<1024, 256, 0, stream>>>((const v4f*)W1, W1b, wn4);
  convert_w<<<1024, 256, 0, stream>>>((const v4f*)W2, W2b, wn4);

  const long long n4 = 100000LL * FDIM / 4;            // full-memory copy
  copy_mem<<<8192, 256, 0, stream>>>((const v4f*)mem, (v4f*)out, n4);

  const size_t smem = (size_t)MTILE * XPITCH * 2 + (size_t)MTILE * HPITCH * 2
                    + (size_t)MTILE * sizeof(long long);  // ~198 KB of 320 KB LDS
  mlp_fused<<<NSEL / MTILE, 256, smem, stream>>>(mem, nodes, W1b, b1, W2b, b2, out);
}
